// Attention_23871428231982
// MI455X (gfx1250) — compile-verified, matrix operand order checked by reading, not run on hardware
//
#include <hip/hip_runtime.h>
#include <hip/hip_bf16.h>

typedef __attribute__((ext_vector_type(16))) __bf16 v16bf;
typedef __attribute__((ext_vector_type(8)))  float  v8f;

#define B_   4
#define N_   2048
#define D_   1024
#define H_   16
#define DH_  64
#define NEGMAX (-3.402823466e38f)

// ---------------------------------------------------------------- helpers

static __device__ inline v8f zero8() {
  v8f z;
#pragma unroll
  for (int i = 0; i < 8; ++i) z[i] = 0.0f;
  return z;
}

static __device__ inline v8f wmma_bf16(v16bf a, v16bf b, v8f c) {
  return __builtin_amdgcn_wmma_f32_16x16x32_bf16(
      false, a, false, b, (short)0, c, false, false);
}

// 16x32 bf16 fragment from a row-major tile whose rows are K-contiguous.
// Every WMMA operand (A, B^T, K^T, V^T, P) is stored so each lane reads two
// 16B-contiguous runs -> 2x ds_load_b128.
static __device__ inline v16bf load_frag(const __bf16* t, int ld, int lane) {
  const int m = lane & 15, half = lane >> 4;
  v16bf a;
#pragma unroll
  for (int j = 0; j < 8; ++j) {
    const int kb = ((j & 4) << 2) + half * 8 + ((j & 3) << 1);
    a[2 * j]     = t[m * ld + kb];
    a[2 * j + 1] = t[m * ld + kb + 1];
  }
  return a;
}

// gfx1250 async copy: 16B global -> LDS, tracked by ASYNCcnt.
static __device__ inline void async_copy16(void* lds, const void* g) {
  unsigned off = (unsigned)(unsigned long long)(uintptr_t)lds;
  asm volatile("global_load_async_to_lds_b128 %0, %1, off"
               :: "v"(off), "v"(g) : "memory");
}

static __device__ inline void wait_async() {
#if __has_builtin(__builtin_amdgcn_s_wait_asynccnt)
  __builtin_amdgcn_s_wait_asynccnt(0);
#else
  asm volatile("s_wait_asynccnt 0" ::: "memory");
#endif
}

// ---------------------------------------------------------------- conversions

__global__ void cvt_f32_bf16(const float* __restrict__ in, __bf16* __restrict__ out, int n) {
  int i = blockIdx.x * blockDim.x + threadIdx.x;
  if (i < n) out[i] = (__bf16)in[i];
}

// f32 (R x C) -> bf16 transposed (C x R), 32x32 LDS tiles, block (32,8)
__global__ void cvtT_f32_bf16(const float* __restrict__ in, __bf16* __restrict__ out,
                              int R, int C) {
  __shared__ float t[32][33];
  const int c0 = blockIdx.x * 32, r0 = blockIdx.y * 32;
#pragma unroll
  for (int i = threadIdx.y; i < 32; i += 8)
    t[i][threadIdx.x] = in[(size_t)(r0 + i) * C + c0 + threadIdx.x];
  __syncthreads();
#pragma unroll
  for (int i = threadIdx.y; i < 32; i += 8)
    out[(size_t)(c0 + i) * R + r0 + threadIdx.x] = (__bf16)t[threadIdx.x][i];
}

// bf16 batched transpose: slices of (R x C) -> (C x R), batch = blockIdx.z
__global__ void transpose_bf16(const __bf16* __restrict__ in, __bf16* __restrict__ out,
                               int R, int C) {
  __shared__ __bf16 t[32][33];
  const __bf16* ins = in  + (size_t)blockIdx.z * R * C;
  __bf16*      outs = out + (size_t)blockIdx.z * R * C;
  const int c0 = blockIdx.x * 32, r0 = blockIdx.y * 32;
#pragma unroll
  for (int i = threadIdx.y; i < 32; i += 8)
    t[i][threadIdx.x] = ins[(size_t)(r0 + i) * C + c0 + threadIdx.x];
  __syncthreads();
#pragma unroll
  for (int i = threadIdx.y; i < 32; i += 8)
    outs[(size_t)(c0 + i) * R + r0 + threadIdx.x] = t[threadIdx.x][i];
}

// ---------------------------------------------------------------- GEMM: C = A * B^T
// A: M x K bf16, Bt: N x K bf16, C: M x N f32. M%128==0, N%128==0, K%32==0.
// Double-buffered async-to-LDS pipeline.

#define BM 128
#define BN 128
#define BK 32
#define LDT 40   // BK + 8 pad: 80B rows -> conflict-free b128 frag reads

__global__ __launch_bounds__(256) void gemm_abT(
    const __bf16* __restrict__ A, const __bf16* __restrict__ Bt,
    float* __restrict__ C, int M, int N, int K) {
  __shared__ __bf16 As[2][BM * LDT];
  __shared__ __bf16 Bs[2][BN * LDT];
  const int tid  = threadIdx.x;
  const int lane = tid & 31;
  const int w    = tid >> 5;
  const int wr   = w >> 1;     // 0..3 : 32-row group
  const int wc   = w & 1;      // 0..1 : 64-col group
  const int m0   = blockIdx.y * BM;
  const int n0   = blockIdx.x * BN;

  auto fill = [&](int buf, int k0) {
#pragma unroll
    for (int i = 0; i < 2; ++i) {
      int c = tid + i * 256;
      int row = c >> 2, cb = (c & 3) * 8;
      async_copy16(&As[buf][row * LDT + cb], &A [(size_t)(m0 + row) * K + k0 + cb]);
      async_copy16(&Bs[buf][row * LDT + cb], &Bt[(size_t)(n0 + row) * K + k0 + cb]);
    }
  };

  v8f acc[2][4];
#pragma unroll
  for (int mi = 0; mi < 2; ++mi)
#pragma unroll
    for (int ni = 0; ni < 4; ++ni) acc[mi][ni] = zero8();

  fill(0, 0);
  wait_async();
  __syncthreads();

  int buf = 0;
  for (int k0 = 0; k0 < K; k0 += BK, buf ^= 1) {
    if (k0 + BK < K) fill(buf ^ 1, k0 + BK);   // stream next tile during compute

    v16bf af[2], bf[4];
#pragma unroll
    for (int mi = 0; mi < 2; ++mi)
      af[mi] = load_frag(&As[buf][(wr * 32 + mi * 16) * LDT], LDT, lane);
#pragma unroll
    for (int ni = 0; ni < 4; ++ni)
      bf[ni] = load_frag(&Bs[buf][(wc * 64 + ni * 16) * LDT], LDT, lane);
#pragma unroll
    for (int mi = 0; mi < 2; ++mi)
#pragma unroll
      for (int ni = 0; ni < 4; ++ni)
        acc[mi][ni] = wmma_bf16(af[mi], bf[ni], acc[mi][ni]);

    wait_async();
    __syncthreads();
  }

  const int half = lane >> 4, n = lane & 15;
#pragma unroll
  for (int mi = 0; mi < 2; ++mi)
#pragma unroll
    for (int ni = 0; ni < 4; ++ni)
#pragma unroll
      for (int r = 0; r < 8; ++r)
        C[(size_t)(m0 + wr * 32 + mi * 16 + r + 8 * half) * N +
          n0 + wc * 64 + ni * 16 + n] = acc[mi][ni][r];
}

// ---------------------------------------------------------------- RoPE
// qkv f32 (B,N,3*H*64) -> q/k/v bf16 (B,H,N,64). q pre-scaled by 1/sqrt(64).

__global__ void rope_kernel(const float* __restrict__ qkv,
                            const float* __restrict__ freqs,
                            __bf16* __restrict__ qb, __bf16* __restrict__ kb,
                            __bf16* __restrict__ vb) {
  int tid = blockIdx.x * blockDim.x + threadIdx.x;
  if (tid >= B_ * H_ * N_ * 32) return;
  int d = tid & 31;
  int rest = tid >> 5;
  int n = rest % N_;  rest /= N_;
  int h = rest % H_;
  int b = rest / H_;

  float f1 = freqs[n * DH_ + d];
  float f2 = freqs[n * DH_ + d + 32];
  float c1 = __cosf(f1), s1 = __sinf(f1);
  float c2 = __cosf(f2), s2 = __sinf(f2);

  size_t ibase = ((size_t)b * N_ + n) * (3 * D_) + h * DH_ + d;
  size_t obase = (((size_t)b * H_ + h) * N_ + n) * DH_ + d;

#pragma unroll
  for (int part = 0; part < 3; ++part) {
    float t1 = qkv[ibase + part * D_];
    float t2 = qkv[ibase + part * D_ + 32];
    float o1 = t1 * c1 - t2 * s1;           // rotate_half: [-x2, x1]
    float o2 = t2 * c2 + t1 * s2;
    float sc = (part == 0) ? 0.125f : 1.0f; // fold softmax scale into q
    __bf16* out = (part == 0) ? qb : ((part == 1) ? kb : vb);
    out[obase]      = (__bf16)(o1 * sc);
    out[obase + 32] = (__bf16)(o2 * sc);
  }
}

// ---------------------------------------------------------------- flash attention
// q,k bf16 (B,H,N,64); vT bf16 (B,H,64,N); -> ao bf16 (B,N,H*64)
// Double-buffered async K/V stream, 8 waves/block, 16 q rows/wave.

#define LQK 72   // 64 + 8 pad (144B rows)
#define LVT 40   // 32 + 8 pad (80B rows)
#define LSB 33
#define LPB 40

__global__ __launch_bounds__(256) void flash_attn(
    const __bf16* __restrict__ qg, const __bf16* __restrict__ kg,
    const __bf16* __restrict__ vTg, const unsigned char* __restrict__ mask,
    __bf16* __restrict__ ao) {
  __shared__ __bf16 Qs[128 * LQK];
  __shared__ __bf16 Ks[2][32 * LQK];
  __shared__ __bf16 VsT[2][64 * LVT];
  __shared__ float  Sbuf[8][16 * LSB];
  __shared__ __bf16 Pbuf[8][16 * LPB];
  __shared__ float  rowm[8][16], rowl[8][16], rowa[8][16];

  const int tid  = threadIdx.x;
  const int lane = tid & 31;
  const int w    = tid >> 5;
  const int half = lane >> 4;
  const int nn   = lane & 15;

  const int b  = blockIdx.y / H_;
  const int h  = blockIdx.y % H_;
  const int n0 = blockIdx.x * 128;
  const size_t bh = (size_t)b * H_ + h;

  auto fillkv = [&](int buf, int j0) {
    const __bf16* ksrc = kg  + (bh * N_ + j0) * DH_;
    const __bf16* vsrc = vTg + bh * (size_t)DH_ * N_ + j0;
    int kr = tid >> 3, kc = (tid & 7) * 8;
    async_copy16(&Ks[buf][kr * LQK + kc], &ksrc[kr * DH_ + kc]);
    int vr = tid >> 2, vc = (tid & 3) * 8;
    async_copy16(&VsT[buf][vr * LVT + vc], &vsrc[(size_t)vr * N_ + vc]);
  };

  // prologue: Q block (128 x 64) + first K/V tile, all async
  {
    const __bf16* qsrc = qg + (bh * N_ + n0) * DH_;
#pragma unroll
    for (int i = 0; i < 4; ++i) {
      int c = tid + i * 256;
      int row = c >> 3, cb = (c & 7) * 8;
      async_copy16(&Qs[row * LQK + cb], &qsrc[row * DH_ + cb]);
    }
  }
  fillkv(0, 0);
  wait_async();
  __syncthreads();

  v16bf q_lo = load_frag(&Qs[w * 16 * LQK], LQK, lane);
  v16bf q_hi = load_frag(&Qs[w * 16 * LQK + 32], LQK, lane);

  v8f o[4];
#pragma unroll
  for (int t = 0; t < 4; ++t) o[t] = zero8();
  if (lane < 16) { rowm[w][lane] = NEGMAX; rowl[w][lane] = 0.0f; }

  int buf = 0;
  for (int j0 = 0; j0 < N_; j0 += 32, buf ^= 1) {
    if (j0 + 32 < N_) fillkv(buf ^ 1, j0 + 32);  // stream next K/V during compute

    // S = Q * K^T (16 x 32)
#pragma unroll
    for (int t = 0; t < 2; ++t) {
      v16bf bk0 = load_frag(&Ks[buf][t * 16 * LQK], LQK, lane);
      v16bf bk1 = load_frag(&Ks[buf][t * 16 * LQK + 32], LQK, lane);
      v8f s = zero8();
      s = wmma_bf16(q_lo, bk0, s);
      s = wmma_bf16(q_hi, bk1, s);
      const int col = t * 16 + nn;
      const bool ok = mask[b * N_ + j0 + col] != 0;
#pragma unroll
      for (int r = 0; r < 8; ++r)
        Sbuf[w][(r + 8 * half) * LSB + col] = ok ? s[r] : NEGMAX;
    }

    // online softmax: lanes 0..15 own one row each (DS in-order per wave)
    if (lane < 16) {
      const int m = lane;
      float mx = rowm[w][m];
      const float oldm = mx;
      float sv[32];
#pragma unroll
      for (int c = 0; c < 32; ++c) { sv[c] = Sbuf[w][m * LSB + c]; mx = fmaxf(mx, sv[c]); }
      float alpha = __expf(oldm - mx);
      float sum = 0.0f;
#pragma unroll
      for (int c = 0; c < 32; ++c) {
        float p = __expf(sv[c] - mx);
        sum += p;
        Pbuf[w][m * LPB + c] = (__bf16)p;
      }
      rowl[w][m] = rowl[w][m] * alpha + sum;
      rowm[w][m] = mx;
      rowa[w][m] = alpha;
    }

    float al[8];
#pragma unroll
    for (int r = 0; r < 8; ++r) al[r] = rowa[w][r + 8 * half];
#pragma unroll
    for (int t = 0; t < 4; ++t)
#pragma unroll
      for (int r = 0; r < 8; ++r) o[t][r] *= al[r];

    // O += P * V  (B-fragments from V^T: K-contiguous -> b128 loads)
    v16bf p = load_frag(&Pbuf[w][0], LPB, lane);
#pragma unroll
    for (int t = 0; t < 4; ++t) {
      v16bf bv = load_frag(&VsT[buf][t * 16 * LVT], LVT, lane);
      o[t] = wmma_bf16(p, bv, o[t]);
    }

    wait_async();
    __syncthreads();
  }

  float inv[8];
#pragma unroll
  for (int r = 0; r < 8; ++r) inv[r] = 1.0f / rowl[w][r + 8 * half];
#pragma unroll
  for (int t = 0; t < 4; ++t)
#pragma unroll
    for (int r = 0; r < 8; ++r) {
      int gr = n0 + w * 16 + r + 8 * half;
      ao[((size_t)b * N_ + gr) * D_ + h * DH_ + t * 16 + nn] =
          (__bf16)(o[t][r] * inv[r]);
    }
}

// ---------------------------------------------------------------- launch

extern "C" void kernel_launch(void* const* d_in, const int* in_sizes, int n_in,
                              void* d_out, int out_size, void* d_ws, size_t ws_size,
                              hipStream_t stream) {
  const float*         x     = (const float*)d_in[0];
  const unsigned char* mask  = (const unsigned char*)d_in[1];  // jnp bool -> 1 byte
  const float*         freqs = (const float*)d_in[2];
  const float*         Wqkv  = (const float*)d_in[3];
  const float*         Wout  = (const float*)d_in[4];
  float*               out   = (float*)d_out;
  char*                ws    = (char*)d_ws;

  const size_t BN_ = (size_t)B_ * N_;   // 8192 rows
  size_t off = 0;
  __bf16* xb     = (__bf16*)(ws + off); off += BN_ * D_ * 2;            // 16 MB
  __bf16* WqkvT  = (__bf16*)(ws + off); off += (size_t)D_ * 3 * D_ * 2; //  6 MB (3072 x 1024)
  __bf16* WoutT  = (__bf16*)(ws + off); off += (size_t)D_ * D_ * 2;     //  2 MB (1024 x 1024)
  float*  qkvf   = (float*)(ws + off);  off += BN_ * 3 * D_ * 4;        // 96 MB
  __bf16* qb     = (__bf16*)(ws + off); off += BN_ * D_ * 2;            // 16 MB
  __bf16* kb     = (__bf16*)(ws + off); off += BN_ * D_ * 2;            // 16 MB
  __bf16* vb     = (__bf16*)(ws + off); off += BN_ * D_ * 2;            // 16 MB
  __bf16* vT     = (__bf16*)(ws + off); off += BN_ * D_ * 2;            // 16 MB
  __bf16* aob    = (__bf16*)(ws + off); off += BN_ * D_ * 2;            // 16 MB

  // 1) x -> bf16 ; weights -> transposed bf16 (N x K for C = A*B^T)
  {
    int nx = B_ * N_ * D_;
    cvt_f32_bf16<<<(nx + 255) / 256, 256, 0, stream>>>(x, xb, nx);
    cvtT_f32_bf16<<<dim3(3 * D_ / 32, D_ / 32), dim3(32, 8), 0, stream>>>(
        Wqkv, WqkvT, D_, 3 * D_);
    cvtT_f32_bf16<<<dim3(D_ / 32, D_ / 32), dim3(32, 8), 0, stream>>>(
        Wout, WoutT, D_, D_);
  }

  // 2) qkv = x @ W_qkv   (8192 x 3072 x 1024)
  gemm_abT<<<dim3(3 * D_ / BN, B_ * N_ / BM), 256, 0, stream>>>(
      xb, WqkvT, qkvf, B_ * N_, 3 * D_, D_);

  // 3) RoPE + head layout
  {
    int nt = B_ * H_ * N_ * 32;
    rope_kernel<<<(nt + 255) / 256, 256, 0, stream>>>(qkvf, freqs, qb, kb, vb);
  }

  // 4) v -> v^T per (b,h): (N x 64) -> (64 x N)
  transpose_bf16<<<dim3(DH_ / 32, N_ / 32, B_ * H_), dim3(32, 8), 0, stream>>>(
      vb, vT, N_, DH_);

  // 5) flash attention -> (B, N, H*64) bf16
  flash_attn<<<dim3(N_ / 128, B_ * H_), 256, 0, stream>>>(qb, kb, vT, mask, aob);

  // 6) out = ao @ W_out  (8192 x 1024 x 1024), fp32 result
  gemm_abT<<<dim3(D_ / BN, B_ * N_ / BM), 256, 0, stream>>>(
      aob, WoutT, out, B_ * N_, D_, D_);
}
